// G_Encoder_63728724738088
// MI455X (gfx1250) — matrix operand order repaired and checked by reading, not compiled, not run on hardware
//
#include <hip/hip_runtime.h>
#include <hip/hip_bf16.h>

typedef __attribute__((ext_vector_type(16))) __bf16 v16bf;
typedef __attribute__((ext_vector_type(8)))  __bf16 v8bf;
typedef __attribute__((ext_vector_type(8)))  float  v8f;

#define F_IN 256
#define NHID 128
#define NB   64
#define BN_EPS 1e-5f

// ---------------------------------------------------------------------------
// 0) zero the small accumulator region (colsum, colsq, bprime, pool_s, pool_c)
// ---------------------------------------------------------------------------
__global__ void k_zero(float* p, int n) {
    int i = blockIdx.x * blockDim.x + threadIdx.x;
    if (i < n) p[i] = 0.0f;
}

// ---------------------------------------------------------------------------
// 1) per-feature sum / sum-of-squares (coalesced: lane f reads column f)
// ---------------------------------------------------------------------------
__global__ void k_colstats(const float* __restrict__ x, float* __restrict__ colsum,
                           float* __restrict__ colsq, int N) {
    int f = threadIdx.x;                      // 256 threads
    float s = 0.f, sq = 0.f;
    for (int row = blockIdx.x; row < N; row += gridDim.x) {
        float v = x[(size_t)row * F_IN + f];
        s += v; sq += v * v;
    }
    atomicAdd(&colsum[f], s);
    atomicAdd(&colsq[f], sq);
}

// ---------------------------------------------------------------------------
// 2) fold BN into the GEMM:  W'[k][j] = a[k]*W[k][j] (bf16),  b'[j] = sum_k c[k]*W[k][j]
//    a[k] = gamma*rsqrt(var+eps), c[k] = beta - mu*a[k].  One block, 256 threads.
// ---------------------------------------------------------------------------
__global__ void k_fold(const float* __restrict__ colsum, const float* __restrict__ colsq,
                       const float* __restrict__ gamma, const float* __restrict__ beta,
                       const float* __restrict__ W, __bf16* __restrict__ Wb,
                       float* __restrict__ bprime, int N) {
    __shared__ float a_s[F_IN];
    __shared__ float c_s[F_IN];
    int t = threadIdx.x;                      // 256
    float inv = 1.0f / (float)N;
    float mu  = colsum[t] * inv;
    float var = colsq[t] * inv - mu * mu;
    float av  = gamma[t] * rsqrtf(var + BN_EPS);
    a_s[t] = av;
    c_s[t] = beta[t] - mu * av;
    __syncthreads();
    for (int i = t; i < F_IN * NHID; i += 256) {
        int k = i >> 7;
        Wb[i] = (__bf16)(a_s[k] * W[i]);
    }
    if (t < NHID) {
        float s = 0.f;
        for (int k = 0; k < F_IN; ++k) s += c_s[k] * W[k * NHID + t];
        bprime[t] = s;
    }
}

// ---------------------------------------------------------------------------
// 3) WMMA GEMM: h = x @ W' + b'   (x f32 -> bf16 on the fly, f32 accumulate)
//    Block = 256 threads (8 waves). Wave w owns M-tile (blockIdx*128 + w*16),
//    keeps its 16x256 A strip in 64 VGPRs, loops all 8 N-tiles reading W'
//    from LDS (transposed + xor-swizzled, exactly 64KB).
// ---------------------------------------------------------------------------
__global__ __launch_bounds__(256) void k_gemm_wmma(
        const float* __restrict__ x, const __bf16* __restrict__ Wb,
        const float* __restrict__ bprime, float* __restrict__ h, int N) {
    __shared__ __bf16 BT[NHID * 256];         // [col][k], chunk-swizzled, 64 KB

    int tid = threadIdx.x;
    // stage W' -> LDS transposed; element (k,col) goes to
    //   BT[col*256 + (((k>>4) ^ (col&7)) << 4) + (k & 15)]
    for (int i = tid; i < F_IN * NHID; i += 256) {
        int k = i >> 7, col = i & 127;
        BT[col * 256 + ((((k >> 4) ^ (col & 7)) << 4) | (k & 15))] = Wb[i];
    }
    __syncthreads();

    int wave  = tid >> 5;
    int lane  = tid & 31;
    int lhalf = lane & 15;
    int hi    = lane >> 4;                    // 0: lanes 0-15, 1: lanes 16-31
    int tileM = blockIdx.x * 128 + wave * 16;

    int rowA = tileM + lhalf;                 // A operand: M = lane&15 (both halves)
    if (rowA >= N) rowA = N - 1;              // clamp loads; stores guarded below
    const float4* xr4 = (const float4*)(x + (size_t)rowA * F_IN);

    // A operand registers: lane half hi=0 holds K {0..15}+32*kc,
    // hi=1 holds K {16..31}+32*kc (matches B's contiguous order).
    v16bf A[8];
#pragma unroll
    for (int kc = 0; kc < 8; ++kc) {
        int kA = kc * 32 + hi * 16;
        float4 f0 = xr4[kA >> 2];
        float4 f1 = xr4[(kA >> 2) + 1];
        float4 f2 = xr4[(kA >> 2) + 2];
        float4 f3 = xr4[(kA >> 2) + 3];
        v16bf a;
        a[0] = (__bf16)f0.x; a[1]  = (__bf16)f0.y; a[2]  = (__bf16)f0.z; a[3]  = (__bf16)f0.w;
        a[4] = (__bf16)f1.x; a[5]  = (__bf16)f1.y; a[6]  = (__bf16)f1.z; a[7]  = (__bf16)f1.w;
        a[8] = (__bf16)f2.x; a[9]  = (__bf16)f2.y; a[10] = (__bf16)f2.z; a[11] = (__bf16)f2.w;
        a[12] = (__bf16)f3.x; a[13] = (__bf16)f3.y; a[14] = (__bf16)f3.z; a[15] = (__bf16)f3.w;
        A[kc] = a;
    }

#pragma unroll
    for (int nt = 0; nt < 8; ++nt) {
        int col = nt * 16 + lhalf;            // B/C/D operand column
        v8f acc = {};
#pragma unroll
        for (int kc = 0; kc < 8; ++kc) {
            // B operand: lanes 0-15 cover K kB..kB+15 (kB=kc*32), lanes 16-31 +16
            int kB = kc * 32 + hi * 16;
            int chunk = ((kB >> 4) ^ (col & 7));
            const v8bf* bp = (const v8bf*)(&BT[col * 256 + (chunk << 4)]);
            v8bf lo = bp[0];
            v8bf hi8 = bp[1];
            v16bf b = __builtin_shufflevector(lo, hi8, 0, 1, 2, 3, 4, 5, 6, 7,
                                              8, 9, 10, 11, 12, 13, 14, 15);
            acc = __builtin_amdgcn_wmma_f32_16x16x32_bf16(
                false, A[kc], false, b, (short)0, acc, false, false);
        }
        float bp2 = bprime[col];
#pragma unroll
        for (int r = 0; r < 8; ++r) {         // D: VGPR r -> M = hi*8 + r
            int row = tileM + hi * 8 + r;
            if (row < N) h[(size_t)row * NHID + col] = acc[r] + bp2;
        }
    }
}

// ---------------------------------------------------------------------------
// 4) degree (incl. self-loop) -> dinv = rsqrt(max(deg,1)), in place
// ---------------------------------------------------------------------------
__global__ void k_deg_init(float* dinv, int N) {
    int i = blockIdx.x * blockDim.x + threadIdx.x;
    if (i < N) dinv[i] = 1.0f;                // self-loop
}
__global__ void k_deg_acc(const long long* __restrict__ dst, float* dinv, int E) {
    int e = blockIdx.x * blockDim.x + threadIdx.x;
    if (e < E) atomicAdd(&dinv[dst[e]], 1.0f);
}
__global__ void k_deg_fin(float* dinv, int N) {
    int i = blockIdx.x * blockDim.x + threadIdx.x;
    if (i < N) dinv[i] = rsqrtf(fmaxf(dinv[i], 1.0f));
}

// ---------------------------------------------------------------------------
// 5) agg init: bias + self-loop message  agg[i][j] = b[j] + h[i][j]*dinv[i]^2
// ---------------------------------------------------------------------------
__global__ void k_agg_init(const float* __restrict__ h, const float* __restrict__ dinv,
                           const float* __restrict__ bias, float* __restrict__ agg, int N) {
    size_t idx = (size_t)blockIdx.x * blockDim.x + threadIdx.x;
    if (idx >= (size_t)N * NHID) return;
    int row = (int)(idx >> 7), j = (int)(idx & 127);
    float di = dinv[row];
    agg[idx] = bias[j] + h[idx] * di * di;
}

// ---------------------------------------------------------------------------
// 6) edge scatter: one wave32 per edge, float4 per lane, f32 atomics on dst
// ---------------------------------------------------------------------------
__global__ void k_edge_agg(const long long* __restrict__ src, const long long* __restrict__ dst,
                           const float* __restrict__ h, const float* __restrict__ dinv,
                           float* __restrict__ agg, int E) {
    int wid = (int)(((size_t)blockIdx.x * blockDim.x + threadIdx.x) >> 5);
    if (wid >= E) return;
    int lane = threadIdx.x & 31;
    long long s = src[wid], d = dst[wid];
    float w = dinv[s] * dinv[d];
    float4 hv = ((const float4*)(h + (size_t)s * NHID))[lane];
    float* ap = agg + (size_t)d * NHID + lane * 4;
    atomicAdd(ap + 0, hv.x * w);
    atomicAdd(ap + 1, hv.y * w);
    atomicAdd(ap + 2, hv.z * w);
    atomicAdd(ap + 3, hv.w * w);
}

// ---------------------------------------------------------------------------
// 7) ReLU + segment-sum pool: LDS partials per block (batch is sorted -> few
//    distinct graph ids per 256-row block), then sparse atomics
// ---------------------------------------------------------------------------
__global__ void k_pool(const float* __restrict__ agg, const long long* __restrict__ batch,
                       float* __restrict__ s, float* __restrict__ cnt, int N) {
    __shared__ float sl[NB * NHID];           // 32 KB
    __shared__ float cl[NB];
    int tid = threadIdx.x;                    // 128 threads: thread = column
    for (int i = tid; i < NB * NHID; i += 128) sl[i] = 0.f;
    if (tid < NB) cl[tid] = 0.f;
    __syncthreads();
    int base = blockIdx.x * 256;
    for (int r = 0; r < 256; ++r) {
        int row = base + r;
        if (row >= N) break;
        int b = (int)batch[row];
        float v = agg[(size_t)row * NHID + tid];
        v = v > 0.f ? v : 0.f;                // ReLU
        sl[b * NHID + tid] += v;
        if (tid == 0) cl[b] += 1.f;
    }
    __syncthreads();
    for (int b = 0; b < NB; ++b) {
        float v = sl[b * NHID + tid];
        if (v != 0.f) atomicAdd(&s[b * NHID + tid], v);
    }
    if (tid < NB && cl[tid] != 0.f) atomicAdd(&cnt[tid], cl[tid]);
}

// ---------------------------------------------------------------------------
// 8) finalize: g = s / max(cnt,1); output tuple is (g, g)
// ---------------------------------------------------------------------------
__global__ void k_final(const float* __restrict__ s, const float* __restrict__ cnt,
                        float* __restrict__ out) {
    int t = blockIdx.x * blockDim.x + threadIdx.x;
    if (t >= NB * NHID) return;
    float g = s[t] / fmaxf(cnt[t >> 7], 1.0f);
    out[t] = g;
    out[NB * NHID + t] = g;
}

// ---------------------------------------------------------------------------
static inline int cdiv(long long a, long long b) { return (int)((a + b - 1) / b); }

extern "C" void kernel_launch(void* const* d_in, const int* in_sizes, int n_in,
                              void* d_out, int out_size, void* d_ws, size_t ws_size,
                              hipStream_t stream) {
    const float*     x     = (const float*)d_in[0];
    const long long* ei    = (const long long*)d_in[1];   // int64 [2,E]
    const long long* batch = (const long long*)d_in[2];   // int64 [N] (sorted)
    // d_in[3] = edge_weight: unused by the reference computation
    const float*     gamma = (const float*)d_in[4];
    const float*     beta  = (const float*)d_in[5];
    const float*     W     = (const float*)d_in[6];
    const float*     bias  = (const float*)d_in[7];
    float*           out   = (float*)d_out;

    const int N = in_sizes[0] / F_IN;
    const int E = in_sizes[1] / 2;
    const long long* srcp = ei;
    const long long* dstp = ei + E;

    // ---- workspace layout (~103.5 MB total) ----
    char* ws = (char*)d_ws;
    float* colsum = (float*)ws;               // 256
    float* colsq  = colsum + F_IN;            // 256
    float* bprime = colsq + F_IN;             // 128
    float* pool_s = bprime + NHID;            // 8192
    float* pool_c = pool_s + NB * NHID;       // 64
    size_t off = (size_t)((char*)(pool_c + NB) - ws);
    off = (off + 255) & ~(size_t)255;
    __bf16* Wb = (__bf16*)(ws + off);  off += (size_t)F_IN * NHID * 2;      // 64 KB
    off = (off + 255) & ~(size_t)255;
    float* dinv = (float*)(ws + off);  off += (size_t)N * 4;                // 400 KB
    off = (off + 255) & ~(size_t)255;
    float* h    = (float*)(ws + off);  off += (size_t)N * NHID * 4;         // 51.2 MB
    off = (off + 255) & ~(size_t)255;
    float* agg  = (float*)(ws + off);                                       // 51.2 MB

    const int zn = F_IN + F_IN + NHID + NB * NHID + NB;  // 8896 floats
    k_zero<<<cdiv(zn, 256), 256, 0, stream>>>(colsum, zn);

    // BN stats + fold into GEMM weights
    k_colstats<<<512, F_IN, 0, stream>>>(x, colsum, colsq, N);
    k_fold<<<1, F_IN, 0, stream>>>(colsum, colsq, gamma, beta, W, Wb, bprime, N);

    // degree chain (independent of GEMM)
    k_deg_init<<<cdiv(N, 256), 256, 0, stream>>>(dinv, N);
    k_deg_acc<<<cdiv(E, 256), 256, 0, stream>>>(dstp, dinv, E);
    k_deg_fin<<<cdiv(N, 256), 256, 0, stream>>>(dinv, N);

    // h = x @ W' + b'  (BN folded in; v_wmma_f32_16x16x32_bf16)
    k_gemm_wmma<<<cdiv(N, 128), 256, 0, stream>>>(x, Wb, bprime, h, N);

    // agg = b + self-loop + edge messages
    k_agg_init<<<cdiv((long long)N * NHID, 256), 256, 0, stream>>>(h, dinv, bias, agg, N);
    k_edge_agg<<<cdiv((long long)E * 32, 256), 256, 0, stream>>>(srcp, dstp, h, dinv, agg, E);

    // ReLU + mean pool, duplicate output tuple
    k_pool<<<cdiv(N, 256), 128, 0, stream>>>(agg, batch, pool_s, pool_c, N);
    k_final<<<cdiv(NB * NHID, 256), 256, 0, stream>>>(pool_s, pool_c, out);
}
